// Decoder_27951647163038
// MI455X (gfx1250) — matrix operand order repaired and checked by reading, not compiled, not run on hardware
//
#include <hip/hip_runtime.h>
#include <hip/hip_bf16.h>

// ---------------------------------------------------------------------------
// Shapes (fixed by the reference)
// ---------------------------------------------------------------------------
#define H 512
#define O 256
#define B 64
#define S 1024
#define T 256
#define KCAT 1280          // H + O + H  (x = [context | target_t | h])
#define NGATES 2048        // 4*H

typedef __attribute__((ext_vector_type(16))) __bf16 v16bf;
typedef __attribute__((ext_vector_type(8)))  float  v8f;

union TileAB { v16bf v; uint4 q[2]; };

// round-to-nearest-even f32 -> bf16 bits
static __device__ __forceinline__ unsigned short f2bf(float f) {
    unsigned int u = __float_as_uint(f);
    u += 0x7FFFu + ((u >> 16) & 1u);
    return (unsigned short)(u >> 16);
}

static __device__ __forceinline__ float sigmoidf_(float x) {
    return 1.0f / (1.0f + __expf(-x));
}

// ---------------------------------------------------------------------------
// Prep kernel 1: Wcat[n][k] (bf16), n in [0,2048), k in [0,1280)
//   k <  768  -> W_ih[n][k]           (W_ih is [2048, 768] row-major)
//   k >= 768  -> W_hh[n][k-768]       (W_hh is [2048, 512] row-major)
// ---------------------------------------------------------------------------
__global__ void prep_wcat_kernel(const float* __restrict__ W_ih,
                                 const float* __restrict__ W_hh,
                                 unsigned short* __restrict__ wcat) {
    int idx = blockIdx.x * blockDim.x + threadIdx.x;   // < 2048*1280
    int n = idx / KCAT;
    int k = idx - n * KCAT;
    float v = (k < (H + O)) ? W_ih[n * (H + O) + k]
                            : W_hh[n * H + (k - (H + O))];
    wcat[idx] = f2bf(v);
}

// ---------------------------------------------------------------------------
// Prep kernel 2: W_fc -> bf16, bias = b_ih + b_hh, copy h0/c0 into state bufs
// index space: [0,131072) Wfc | [131072,133120) bias | h copy | c copy
// ---------------------------------------------------------------------------
__global__ void prep_misc_kernel(const float* __restrict__ W_fc,
                                 const float* __restrict__ b_ih,
                                 const float* __restrict__ b_hh,
                                 const float* __restrict__ hidden,
                                 const float* __restrict__ cell,
                                 unsigned short* __restrict__ wfc,
                                 float* __restrict__ bias,
                                 float* __restrict__ h_buf,
                                 float* __restrict__ c_buf) {
    int idx = blockIdx.x * blockDim.x + threadIdx.x;   // < 198656
    if (idx < O * H) {
        wfc[idx] = f2bf(W_fc[idx]);
    } else if (idx < O * H + NGATES) {
        int j = idx - O * H;
        bias[j] = b_ih[j] + b_hh[j];
    } else if (idx < O * H + NGATES + B * H) {
        int j = idx - (O * H + NGATES);
        h_buf[j] = hidden[j];                          // hidden is (1,B,H) flat
    } else {
        int j = idx - (O * H + NGATES + B * H);
        c_buf[j] = cell[j];
    }
}

// ---------------------------------------------------------------------------
// Attention + x-assembly. One block per batch element, 256 threads (8 waves).
//   scores[s] = dot(enc[b,s,:], h[b,:])  (fp32, wave-parallel + shfl reduce)
//   softmax over S in LDS
//   context[k] = sum_s p[s] * enc[b,s,k]
//   x_bf[b] = bf16([context | target[b,t,:] | h[b,:]])   (K = 1280)
// enc stays fp32: 128MB fits in the 192MB L2, re-reads are L2 hits.
// ---------------------------------------------------------------------------
__global__ void attn_kernel(const float* __restrict__ enc,
                            const float* __restrict__ h_buf,
                            const float* __restrict__ target,
                            unsigned short* __restrict__ xbf,
                            int t) {
    __shared__ float sh_h[H];
    __shared__ float sh_s[S];
    __shared__ float sh_red[256];

    const int b   = blockIdx.x;
    const int tid = threadIdx.x;
    const float* encB = enc + (size_t)b * S * H;

    sh_h[tid]       = h_buf[b * H + tid];
    sh_h[tid + 256] = h_buf[b * H + tid + 256];
    __syncthreads();

    const int wave = tid >> 5;
    const int lane = tid & 31;

    // hoist this lane's h slice into registers (each lane owns 16 h-dims)
    float hreg[16];
#pragma unroll
    for (int j = 0; j < 16; ++j) hreg[j] = sh_h[lane * 16 + j];

    // ---- scores ----
    for (int s = wave; s < S; s += 8) {
        const float* row = encB + (size_t)s * H + lane * 16;
        float p = 0.0f;
#pragma unroll
        for (int j = 0; j < 16; ++j) p += row[j] * hreg[j];
#pragma unroll
        for (int off = 16; off > 0; off >>= 1) p += __shfl_down(p, off, 32);
        if (lane == 0) sh_s[s] = p;
    }
    __syncthreads();

    // ---- softmax: max ----
    float m = -3.402823466e38f;
#pragma unroll
    for (int i = 0; i < 4; ++i) m = fmaxf(m, sh_s[tid + i * 256]);
    sh_red[tid] = m;
    __syncthreads();
    for (int st = 128; st > 0; st >>= 1) {
        if (tid < st) sh_red[tid] = fmaxf(sh_red[tid], sh_red[tid + st]);
        __syncthreads();
    }
    const float mx = sh_red[0];
    __syncthreads();

    // ---- softmax: exp + sum ----
    float lsum = 0.0f;
#pragma unroll
    for (int i = 0; i < 4; ++i) {
        float e = __expf(sh_s[tid + i * 256] - mx);
        sh_s[tid + i * 256] = e;
        lsum += e;
    }
    sh_red[tid] = lsum;
    __syncthreads();
    for (int st = 128; st > 0; st >>= 1) {
        if (tid < st) sh_red[tid] += sh_red[tid + st];
        __syncthreads();
    }
    const float inv = 1.0f / sh_red[0];

    // ---- context (each thread owns k = tid and k = tid+256) ----
    float acc0 = 0.0f, acc1 = 0.0f;
    for (int s = 0; s < S; ++s) {
        const float p = sh_s[s];
        acc0 += p * encB[(size_t)s * H + tid];
        acc1 += p * encB[(size_t)s * H + tid + 256];
    }

    // ---- assemble bf16 A-matrix row x = [context | target_t | h] ----
    unsigned short* xr = xbf + b * KCAT;
    xr[tid]        = f2bf(acc0 * inv);
    xr[tid + 256]  = f2bf(acc1 * inv);
    xr[512 + tid]  = f2bf(target[((size_t)b * T + t) * O + tid]);   // O == 256
    xr[768 + tid]  = f2bf(sh_h[tid]);
    xr[1024 + tid] = f2bf(sh_h[tid + 256]);
}

// ---------------------------------------------------------------------------
// Fused gates GEMM + LSTM cell update.
// gates = x @ Wcat^T + bias, shape [64, 2048]; Wcat is [2048, 1280] bf16.
// grid = (32 j-tiles, 4 m-tiles), block = 128 threads = 4 waves.
// Wave w computes gate w's 16x16 tile at n = w*512 + jtile*16, accumulating
// over K=1280 with v_wmma_f32_16x16x32_bf16. The four gate tiles for the same
// (batch-tile, j-tile) then meet in LDS so the cell update is local.
//
// WMMA operand layouts (ISA 7.12.2, 16-bit, wave32):
//   A 16x32: lane L holds row (L&15); K-chunks {h*8..h*8+7, 16+h*8..} (h=L>>4)
//   B 32x16: lane L holds col (L&15); K-chunks {h*8..h*8+7, 16+h*8..} (h=L>>4)
// Row-major [N][K] weights give two 16B chunks per lane: no transpose needed.
// ---------------------------------------------------------------------------
__global__ void gates_kernel(const unsigned short* __restrict__ xbf,
                             const unsigned short* __restrict__ wcat,
                             const float* __restrict__ bias,
                             float* __restrict__ c_buf,
                             float* __restrict__ h_buf,
                             unsigned short* __restrict__ hbf) {
    __shared__ float lds_g[4][16][16];

    const int jtile = blockIdx.x;          // 0..31
    const int mtile = blockIdx.y;          // 0..3
    const int tid   = threadIdx.x;
    const int wave  = tid >> 5;            // gate index: 0=i 1=f 2=g 3=o
    const int lane  = tid & 31;
    const int r     = lane & 15;
    const int half  = lane >> 4;

    const int arow = mtile * 16 + r;                   // A row (batch)
    const int ncol = wave * H + jtile * 16 + r;        // B row in Wcat = output n

    const unsigned short* aptr = xbf  + (size_t)arow * KCAT;
    const unsigned short* bptr = wcat + (size_t)ncol * KCAT;

    v8f acc = {0.f, 0.f, 0.f, 0.f, 0.f, 0.f, 0.f, 0.f};

    for (int kt = 0; kt < KCAT / 32; ++kt) {
        TileAB a, bb;
        const int ka = kt * 32 + half * 8;
        a.q[0]  = *(const uint4*)(aptr + ka);
        a.q[1]  = *(const uint4*)(aptr + ka + 16);
        const int kb = kt * 32 + half * 8;
        bb.q[0] = *(const uint4*)(bptr + kb);
        bb.q[1] = *(const uint4*)(bptr + kb + 16);
        acc = __builtin_amdgcn_wmma_f32_16x16x32_bf16(
            false, a.v, false, bb.v, (short)0, acc, false, false);
    }

    // D layout: VGPR v, lane L -> M = v + 8*(L>>4), N = L&15
#pragma unroll
    for (int v = 0; v < 8; ++v) lds_g[wave][v + 8 * half][r] = acc[v];
    __syncthreads();

    // LSTM cell update: 256 tile elements, 128 threads -> 2 each
    for (int e = tid; e < 256; e += 128) {
        const int rr = e >> 4;             // batch row within tile
        const int cc = e & 15;             // j col within tile
        const int bidx = mtile * 16 + rr;
        const int j    = jtile * 16 + cc;
        float ig = lds_g[0][rr][cc] + bias[j];
        float fg = lds_g[1][rr][cc] + bias[H + j];
        float gg = lds_g[2][rr][cc] + bias[2 * H + j];
        float og = lds_g[3][rr][cc] + bias[3 * H + j];
        ig = sigmoidf_(ig);
        fg = sigmoidf_(fg);
        gg = tanhf(gg);
        og = sigmoidf_(og);
        const float cn = fg * c_buf[bidx * H + j] + ig * gg;
        const float hn = og * tanhf(cn);
        c_buf[bidx * H + j] = cn;
        h_buf[bidx * H + j] = hn;
        hbf[bidx * H + j]   = f2bf(hn);
    }
}

// ---------------------------------------------------------------------------
// FC GEMM: out[:, t, :] = h_new @ W_fc^T + b_fc.   [64,512]x[512,256]
// 64 output tiles; 16 blocks x 4 waves, one 16x16 tile per wave, K=512.
// ---------------------------------------------------------------------------
__global__ void fc_kernel(const unsigned short* __restrict__ hbf,
                          const unsigned short* __restrict__ wfc,
                          const float* __restrict__ b_fc,
                          float* __restrict__ out,
                          int t) {
    const int tid  = threadIdx.x;
    const int wave = tid >> 5;
    const int lane = tid & 31;
    const int id   = blockIdx.x * 4 + wave;    // 0..63
    const int mt   = id >> 4;                  // 0..3
    const int nt   = id & 15;                  // 0..15
    const int r    = lane & 15;
    const int half = lane >> 4;

    const unsigned short* aptr = hbf + (size_t)(mt * 16 + r) * H;
    const unsigned short* bptr = wfc + (size_t)(nt * 16 + r) * H;

    v8f acc = {0.f, 0.f, 0.f, 0.f, 0.f, 0.f, 0.f, 0.f};

    for (int kt = 0; kt < H / 32; ++kt) {
        TileAB a, bb;
        const int ka = kt * 32 + half * 8;
        a.q[0]  = *(const uint4*)(aptr + ka);
        a.q[1]  = *(const uint4*)(aptr + ka + 16);
        const int kb = kt * 32 + half * 8;
        bb.q[0] = *(const uint4*)(bptr + kb);
        bb.q[1] = *(const uint4*)(bptr + kb + 16);
        acc = __builtin_amdgcn_wmma_f32_16x16x32_bf16(
            false, a.v, false, bb.v, (short)0, acc, false, false);
    }

#pragma unroll
    for (int v = 0; v < 8; ++v) {
        const int bidx = mt * 16 + v + 8 * half;
        const int o    = nt * 16 + r;
        out[((size_t)bidx * T + t) * O + o] = acc[v] + b_fc[o];
    }
}

// ---------------------------------------------------------------------------
// Host driver: prep once, then T steps of {attention, gates+cell, fc}.
// All launches on `stream` (graph-capture safe). Workspace layout (bytes):
//   Wcat   2048*1280*2 = 5,242,880
//   Wfc     256* 512*2 =   262,144
//   bias        2048*4 =     8,192
//   h_buf    64* 512*4 =   131,072
//   c_buf    64* 512*4 =   131,072
//   hbf      64* 512*2 =    65,536
//   x_bf     64*1280*2 =   163,840   -> ~6.0 MB total
// ---------------------------------------------------------------------------
extern "C" void kernel_launch(void* const* d_in, const int* in_sizes, int n_in,
                              void* d_out, int out_size, void* d_ws, size_t ws_size,
                              hipStream_t stream) {
    const float* enc    = (const float*)d_in[0];
    const float* hidden = (const float*)d_in[1];
    const float* cell   = (const float*)d_in[2];
    const float* target = (const float*)d_in[3];
    const float* W_ih   = (const float*)d_in[4];
    const float* W_hh   = (const float*)d_in[5];
    const float* b_ih   = (const float*)d_in[6];
    const float* b_hh   = (const float*)d_in[7];
    const float* W_fc   = (const float*)d_in[8];
    const float* b_fc   = (const float*)d_in[9];
    float* out = (float*)d_out;

    char* ws = (char*)d_ws;
    size_t off = 0;
    unsigned short* wcat = (unsigned short*)(ws + off); off += (size_t)NGATES * KCAT * 2;
    unsigned short* wfc  = (unsigned short*)(ws + off); off += (size_t)O * H * 2;
    float* bias          = (float*)(ws + off);          off += (size_t)NGATES * 4;
    float* h_buf         = (float*)(ws + off);          off += (size_t)B * H * 4;
    float* c_buf         = (float*)(ws + off);          off += (size_t)B * H * 4;
    unsigned short* hbf  = (unsigned short*)(ws + off); off += (size_t)B * H * 2;
    unsigned short* xbf  = (unsigned short*)(ws + off); off += (size_t)B * KCAT * 2;

    // one-time prep
    prep_wcat_kernel<<<(NGATES * KCAT) / 256, 256, 0, stream>>>(W_ih, W_hh, wcat);
    const int misc_n = O * H + NGATES + 2 * B * H;  // 198656, divisible by 256
    prep_misc_kernel<<<misc_n / 256, 256, 0, stream>>>(
        W_fc, b_ih, b_hh, hidden, cell, wfc, bias, h_buf, c_buf);

    // recurrent decode
    for (int t = 0; t < T; ++t) {
        attn_kernel<<<B, 256, 0, stream>>>(enc, h_buf, target, xbf, t);
        gates_kernel<<<dim3(H / 16, B / 16), 128, 0, stream>>>(
            xbf, wcat, bias, c_buf, h_buf, hbf);
        fc_kernel<<<(B / 16) * (O / 16) / 4, 128, 0, stream>>>(
            hbf, wfc, b_fc, out, t);
    }
}
